// ScaledDotProductAttention_73607149519013
// MI455X (gfx1250) — matrix operand order repaired and checked by reading, not compiled, not run on hardware
//
#include <hip/hip_runtime.h>
#include <hip/hip_bf16.h>
#include <stdint.h>
#include <stddef.h>

typedef __bf16 bf16_t;
typedef __attribute__((ext_vector_type(16))) __bf16 v16bf;
typedef __attribute__((ext_vector_type(8)))  float  v8f;
typedef __attribute__((ext_vector_type(4)))  float  f32x4;

#define SQ    2048
#define DD    64
#define PP    2052   // P-tile row pitch (floats); 2052 % 64 == 4 -> bank stagger, conflict-free column gathers
#define VP    68     // V-chunk row pitch (floats); multiple of 4 keeps 16B alignment for b128
#define NTHR  128
#define SMEM_FLOATS (16 * PP + 32 * VP + 16 * 9 + 16 + 16)

// ---- CDNA5 async memory->LDS copy (ISA 10.7 / 15.18.3 op 98, tracked by ASYNCcnt) ----
// Generic __shared__ pointers carry the LDS offset in their low 32 bits (ISA 10.2),
// so truncation yields the LDS byte address for VDST.
__device__ __forceinline__ void async_copy_b128(void* lds_dst, const void* gsrc) {
  uint32_t lds_addr = (uint32_t)(uintptr_t)lds_dst;
  asm volatile("global_load_async_to_lds_b128 %0, %1, off"
               :: "v"(lds_addr), "v"(gsrc)
               : "memory");
}
__device__ __forceinline__ void wait_asynccnt0() {
  asm volatile("s_wait_asynccnt 0x0" ::: "memory");
}

// A-operand fill for V_WMMA_F32_16X16X32_BF16 (16-bit A 16x32, contiguous K order, not the ISA table order):
// lane holds M = lane&15; elements j=0..15 -> K = base+j
// caller passes base already offset by (lane<16 ? 0 : 16).
__device__ __forceinline__ void fill_a(v16bf& dst, const float* base) {
  f32x4 u0 = *(const f32x4*)(base);
  f32x4 u1 = *(const f32x4*)(base + 4);
  f32x4 u2 = *(const f32x4*)(base + 8);
  f32x4 u3 = *(const f32x4*)(base + 12);
  dst[0]  = (bf16_t)u0.x; dst[1]  = (bf16_t)u0.y; dst[2]  = (bf16_t)u0.z; dst[3]  = (bf16_t)u0.w;
  dst[4]  = (bf16_t)u1.x; dst[5]  = (bf16_t)u1.y; dst[6]  = (bf16_t)u1.z; dst[7]  = (bf16_t)u1.w;
  dst[8]  = (bf16_t)u2.x; dst[9]  = (bf16_t)u2.y; dst[10] = (bf16_t)u2.z; dst[11] = (bf16_t)u2.w;
  dst[12] = (bf16_t)u3.x; dst[13] = (bf16_t)u3.y; dst[14] = (bf16_t)u3.z; dst[15] = (bf16_t)u3.w;
}

// B-operand fill from 16 contiguous floats (16-bit B 32x16, contiguous K order, not the ISA table order):
// lane holds N = lane&15; elements j=0..15 -> K = (lane<16?0:16) + j  (caller offsets base).
__device__ __forceinline__ void fill_b16(v16bf& dst, const float* base) {
  f32x4 u0 = *(const f32x4*)(base);
  f32x4 u1 = *(const f32x4*)(base + 4);
  f32x4 u2 = *(const f32x4*)(base + 8);
  f32x4 u3 = *(const f32x4*)(base + 12);
  dst[0]  = (bf16_t)u0.x; dst[1]  = (bf16_t)u0.y; dst[2]  = (bf16_t)u0.z; dst[3]  = (bf16_t)u0.w;
  dst[4]  = (bf16_t)u1.x; dst[5]  = (bf16_t)u1.y; dst[6]  = (bf16_t)u1.z; dst[7]  = (bf16_t)u1.w;
  dst[8]  = (bf16_t)u2.x; dst[9]  = (bf16_t)u2.y; dst[10] = (bf16_t)u2.z; dst[11] = (bf16_t)u2.w;
  dst[12] = (bf16_t)u3.x; dst[13] = (bf16_t)u3.y; dst[14] = (bf16_t)u3.z; dst[15] = (bf16_t)u3.w;
}

__global__ __launch_bounds__(NTHR) void fa_fwd_attn_kernel(
    const float* __restrict__ qg, const float* __restrict__ kg,
    const float* __restrict__ vg, float* __restrict__ outg,
    float* __restrict__ attng)
{
  extern __shared__ float smem[];
  float* lds_p  = smem;                 // 16 x PP  unnormalized softmax numerators (f32)
  float* lds_v  = lds_p + 16 * PP;      // 32 x VP  V chunk
  float* red    = lds_v + 32 * VP;      // 16 x 9   row partials (padded)
  float* rowmax = red + 16 * 9;         // 16
  float* rowinv = rowmax + 16;          // 16

  const int tid  = threadIdx.x;
  const int wave = tid >> 5;
  const int lane = tid & 31;

  const int bh   = blockIdx.x >> 7;     // blockIdx / (SQ/16)
  const int tile = blockIdx.x & 127;
  const int q0   = tile << 4;
  const int nkb  = tile + 1;            // causal: only k-blocks 0..tile are live

  const float* qbh = qg + (size_t)bh * SQ * DD;
  const float* kbh = kg + (size_t)bh * SQ * DD;
  const float* vbh = vg + (size_t)bh * SQ * DD;
  float* obh = outg  + (size_t)bh * SQ * DD;
  float* abh = attng + (size_t)bh * SQ * SQ;

  const int n     = lane & 15;               // B-operand column / C column
  const int M     = lane & 15;               // A-operand row
  const int bK8   = (lane < 16) ? 0 : 8;     // unused; A and B both offset K by bK16
  const int bK16  = (lane < 16) ? 0 : 16;    // B-operand K offset
  const int Mbase = (lane < 16) ? 0 : 8;     // C/D row base

  // ---------------- Pass 1: S = (Q K^T)/8 + causal mask, into LDS ----------------
  v16bf aq0, aq1;
  {
    const float* qrow = qbh + (size_t)(q0 + M) * DD;
    fill_a(aq0, qrow + bK16);        // d = 0..31 half
    fill_a(aq1, qrow + 32 + bK16);   // d = 32..63 half
  }

  for (int kb = wave; kb < nkb; kb += 4) {
    const float* krow = kbh + (size_t)(kb * 16 + n) * DD + bK16;
    if (kb + 4 < nkb)
      __builtin_prefetch(kbh + (size_t)((kb + 4) * 16 + n) * DD + bK16, 0, 1);
    v16bf bm0, bm1;
    fill_b16(bm0, krow);            // d = 0..31 half
    fill_b16(bm1, krow + 32);       // d = 32..63 half
    v8f c = {};
    c = __builtin_amdgcn_wmma_f32_16x16x32_bf16(false, aq0, false, bm0, (short)0, c, false, false);
    c = __builtin_amdgcn_wmma_f32_16x16x32_bf16(false, aq1, false, bm1, (short)0, c, false, false);
    const int gk = kb * 16 + n;
#pragma unroll
    for (int r = 0; r < 8; ++r) {
      const int lr = Mbase + r;
      float val = c[r] * 0.125f + ((gk > q0 + lr) ? -1.0e9f : 0.0f);
      lds_p[lr * PP + gk] = val;
    }
  }

  const int ncols = nkb * 16;
  const int nc32  = (ncols + 31) & ~31;
  if (nc32 > ncols) {                  // zero-pad one 16-col block so PV runs in 32-wide steps
    for (int i = tid; i < 256; i += NTHR)
      lds_p[(i >> 4) * PP + ncols + (i & 15)] = 0.0f;
  }
  __syncthreads();

  // ---------------- Softmax over live columns (8 threads per row) ----------------
  const int srow = tid >> 3, ssub = tid & 7;
  {
    float mx = -3.402823e38f;
    for (int c0 = ssub * 4; c0 < ncols; c0 += 32) {
      f32x4 t = *(const f32x4*)(lds_p + srow * PP + c0);
      mx = fmaxf(mx, fmaxf(fmaxf(t.x, t.y), fmaxf(t.z, t.w)));
    }
    red[srow * 9 + ssub] = mx;
  }
  __syncthreads();
  if (tid < 16) {
    float m = red[tid * 9];
#pragma unroll
    for (int j = 1; j < 8; ++j) m = fmaxf(m, red[tid * 9 + j]);
    rowmax[tid] = m;
  }
  __syncthreads();
  {
    const float rm = rowmax[srow];
    float sum = 0.0f;
    for (int c0 = ssub * 4; c0 < ncols; c0 += 32) {
      float* p = lds_p + srow * PP + c0;
      f32x4 t = *(const f32x4*)p;
      t.x = __expf(t.x - rm); t.y = __expf(t.y - rm);
      t.z = __expf(t.z - rm); t.w = __expf(t.w - rm);
      sum += (t.x + t.y) + (t.z + t.w);
      *(f32x4*)p = t;
    }
    red[srow * 9 + ssub] = sum;
  }
  __syncthreads();
  if (tid < 16) {
    float s = 0.0f;
#pragma unroll
    for (int j = 0; j < 8; ++j) s += red[tid * 9 + j];
    rowinv[tid] = 1.0f / s;
  }
  __syncthreads();

  // ---------------- attn store: full 16 x 2048 rows, coalesced float4 ----------------
  {
    float* arow = abh + (size_t)q0 * SQ;
    for (int i4 = tid * 4; i4 < 16 * SQ; i4 += NTHR * 4) {
      const int r = i4 >> 11, c = i4 & (SQ - 1);
      f32x4 t;
      if (c < ncols) {
        t = *(const f32x4*)(lds_p + r * PP + c);
        const float inv = rowinv[r];
        t.x *= inv; t.y *= inv; t.z *= inv; t.w *= inv;
      } else {
        t.x = 0.0f; t.y = 0.0f; t.z = 0.0f; t.w = 0.0f;
      }
      *(f32x4*)(arow + (size_t)r * SQ + c) = t;
    }
  }

  // ---------------- Pass 2: out = (P_e . V) * rowinv ----------------
  v8f acc = {};
  for (int kc = 0; kc < nc32; kc += 32) {
    __syncthreads();                   // previous chunk fully consumed
    // async stage V[kc..kc+31][0..63] into LDS (memory -> LDS, no VGPR round-trip)
    for (int i4 = tid * 4; i4 < 32 * DD; i4 += NTHR * 4) {
      const int r = i4 >> 6, c = i4 & 63;
      async_copy_b128(lds_v + r * VP + c, vbh + (size_t)(kc + r) * DD + c);
    }
    if (kc + 32 < nc32)
      __builtin_prefetch(vbh + (size_t)(kc + 32) * DD + tid * 16, 0, 1);
    wait_asynccnt0();                  // this wave's async copies landed in LDS
    __syncthreads();                   // all waves' copies visible

    v16bf am, bm;
    fill_a(am, lds_p + M * PP + kc + bK16);               // P rows, K runs per A layout
    {
      const float* p0 = lds_v + bK16 * VP + wave * 16 + n;  // V[k, d] column gather per B layout
#pragma unroll
      for (int j = 0; j < 16; ++j) bm[j] = (bf16_t)p0[j * VP];
    }
    acc = __builtin_amdgcn_wmma_f32_16x16x32_bf16(false, am, false, bm, (short)0, acc, false, false);
  }

  {
    float* obase = obh + (size_t)(q0 + Mbase) * DD + wave * 16 + n;
#pragma unroll
    for (int r = 0; r < 8; ++r)
      obase[(size_t)r * DD] = acc[r] * rowinv[Mbase + r];
  }
}

extern "C" void kernel_launch(void* const* d_in, const int* in_sizes, int n_in,
                              void* d_out, int out_size, void* d_ws, size_t ws_size,
                              hipStream_t stream) {
  const float* q = (const float*)d_in[0];
  const float* k = (const float*)d_in[1];
  const float* v = (const float*)d_in[2];
  // d_in[3] is the causal mask (1 - tril); applied analytically inside the kernel.
  float* out = (float*)d_out;
  const int BH = in_sizes[0] / (SQ * DD);          // B*H = 32
  float* attn = out + (size_t)BH * SQ * DD;        // outputs concatenated: out | attn

  const dim3 grid(BH * (SQ / 16));                 // 4096 workgroups
  const size_t smem_bytes = (size_t)SMEM_FLOATS * sizeof(float);  // ~137.4 KB
  fa_fwd_attn_kernel<<<grid, NTHR, smem_bytes, stream>>>(q, k, v, out, attn);
}
